// ApertisAttention_43439299232049
// MI455X (gfx1250) — compile-verified, matrix operand order checked by reading, not run on hardware
//
#include <hip/hip_runtime.h>
#include <stdint.h>

typedef __attribute__((ext_vector_type(16))) __bf16 v16bf;
typedef __attribute__((ext_vector_type(8)))  __bf16 v8bf;
typedef __attribute__((ext_vector_type(8)))  float  v8f;

// Problem constants
constexpr int Bc  = 2;
constexpr int Sc  = 2048;
constexpr int Hc  = 1024;
constexpr int NHc = 16;
constexpr int HDc = 64;
constexpr int WIN = 1024;
constexpr int Mrows = Bc * Sc;           // 4096 tokens
constexpr float SCALE = 0.125f;          // 1/sqrt(64)

#define DEVFN __device__ __forceinline__

// ---------------------------------------------------------------------------
// WMMA helpers (CDNA5 wave32, 16x16x32 BF16, f32 accumulate)
//
// A-frag (16x32, row-major source): lane L holds row m = L%16; element e holds
//   K = e + 8*(e>=8) + 8*(L>=16)  -> two contiguous 8-elem (16B) chunks.
// B-frag (32x16, B[k][n]): lane L holds column n = L%16 of B, i.e. row n of the
//   K-contiguous operand, same K mapping (mirror of A).
// C/D frag (v8f): element r is (M = r + 8*(L>=16), N = L%16).
// ---------------------------------------------------------------------------
DEVFN v16bf load_frag16(const __bf16* p /* row base + hi8 */, int kk) {
  v8bf lo = *(const v8bf*)(p + kk);
  v8bf hi = *(const v8bf*)(p + kk + 16);
  v16bf f;
#pragma unroll
  for (int j = 0; j < 8; ++j) { f[j] = lo[j]; f[j + 8] = hi[j]; }
  return f;
}

DEVFN v8f wmma_bf16(v16bf a, v16bf b, v8f c) {
  return __builtin_amdgcn_wmma_f32_16x16x32_bf16(
      /*neg_a=*/false, a, /*neg_b=*/false, b,
      /*c_mod=*/(short)0, c, /*reuse_a=*/false, /*reuse_b=*/false);
}

// ---------------------------------------------------------------------------
// Kernel 0: f32 -> bf16 conversion
// ---------------------------------------------------------------------------
__global__ void cvt_f32_bf16_kernel(const float* __restrict__ x,
                                    __bf16* __restrict__ y, int n) {
  int i = blockIdx.x * blockDim.x + threadIdx.x;
  if (i < n) y[i] = (__bf16)x[i];
}

// ---------------------------------------------------------------------------
// Kernel 1: fused Q/K/V projection. y = x @ W^T + b.
// One wave computes one 16x16 tile; blockIdx.z selects {q,k,v}.
// q,k stored [B][NH][S][HD]; v stored transposed [B][NH][HD][S].
// ---------------------------------------------------------------------------
__global__ __launch_bounds__(32) void qkv_proj_kernel(
    const __bf16* __restrict__ xb,
    const __bf16* __restrict__ wq, const __bf16* __restrict__ wk,
    const __bf16* __restrict__ wv,
    const float* __restrict__ bq, const float* __restrict__ bk,
    const float* __restrict__ bv,
    __bf16* __restrict__ qo, __bf16* __restrict__ ko,
    __bf16* __restrict__ vto) {
  const int lane = threadIdx.x;
  const int lm   = lane & 15;
  const int hi   = lane >> 4;
  const int hi8  = hi << 3;
  const int m0   = blockIdx.x * 16;
  const int n0   = blockIdx.y * 16;
  const int which = blockIdx.z;

  const __bf16* W    = (which == 0) ? wq : (which == 1) ? wk : wv;
  const float*  bias = (which == 0) ? bq : (which == 1) ? bk : bv;

  const __bf16* arow = xb + (size_t)(m0 + lm) * Hc + hi8;
  const __bf16* brow = W  + (size_t)(n0 + lm) * Hc + hi8;

  v8f acc;
#pragma unroll
  for (int r = 0; r < 8; ++r) acc[r] = 0.0f;

#pragma unroll 4
  for (int kk = 0; kk < Hc; kk += 32) {
    v16bf a = load_frag16(arow, kk);
    v16bf b = load_frag16(brow, kk);
    acc = wmma_bf16(a, b, acc);
  }

  const float bn = bias[n0 + lm];
  const int n = n0 + lm;
  const int h = n >> 6;
  const int d = n & 63;
#pragma unroll
  for (int r = 0; r < 8; ++r) {
    const int t  = m0 + r + hi * 8;     // global token row in [0, B*S)
    const int bb = t / Sc;
    const int ss = t - bb * Sc;
    const __bf16 o = (__bf16)(acc[r] + bn);
    if (which == 2) {
      vto[(((size_t)bb * NHc + h) * HDc + d) * Sc + ss] = o;
    } else {
      __bf16* dst = (which == 0) ? qo : ko;
      dst[(((size_t)bb * NHc + h) * Sc + ss) * HDc + d] = o;
    }
  }
}

// ---------------------------------------------------------------------------
// Kernel 2: RoPE applied in-place (f32 math on bf16 storage).
// One thread rotates the (d, d+32) pair. 2^21 threads total.
// ---------------------------------------------------------------------------
__global__ void rope_kernel(__bf16* __restrict__ x,
                            const long long* __restrict__ pos_ids) {
  int gid = blockIdx.x * blockDim.x + threadIdx.x;
  const int j   = gid & 31;          // 5 bits
  int rest      = gid >> 5;
  const int ss  = rest & (Sc - 1);   // 11 bits
  rest >>= 11;
  const int h   = rest & (NHc - 1);  // 4 bits
  const int b   = rest >> 4;         // 1 bit

  long long p = pos_ids[(size_t)b * Sc + ss];
  if (p < 0) p = 0;
  if (p > Sc - 1) p = Sc - 1;

  const float LOG_THETA = 9.210340371976184f;  // ln(10000)
  const float inv = __expf(-(float)j * (LOG_THETA / 32.0f));
  const float ang = (float)p * inv;
  float sn, cs;
  __sincosf(ang, &sn, &cs);

  __bf16* base = x + (((size_t)b * NHc + h) * Sc + ss) * HDc;
  const float x1 = (float)base[j];
  const float x2 = (float)base[j + 32];
  base[j]      = (__bf16)(x1 * cs - x2 * sn);
  base[j + 32] = (__bf16)(x2 * cs + x1 * sn);
}

// ---------------------------------------------------------------------------
// Kernel 3: flash attention with sliding causal window.
// One wave: 16 queries x full head. 32-key blocks, online softmax in f32.
// ---------------------------------------------------------------------------
__global__ __launch_bounds__(32) void attn_kernel(
    const __bf16* __restrict__ qg, const __bf16* __restrict__ kg,
    const __bf16* __restrict__ vtg, __bf16* __restrict__ ctxg) {
  __shared__ __align__(16) __bf16 ldsp[16][32];  // P tile staging (C->A relayout)

  const int lane = threadIdx.x;
  const int lm   = lane & 15;
  const int hi   = lane >> 4;
  const int hi8  = hi << 3;
  const int qb0  = blockIdx.x * 16;
  const int h    = blockIdx.y;
  const int b    = blockIdx.z;

  const __bf16* qbase = qg  + (((size_t)b * NHc + h) * Sc) * HDc;
  const __bf16* kbase = kg  + (((size_t)b * NHc + h) * Sc) * HDc;
  const __bf16* vbase = vtg + (((size_t)b * NHc + h) * HDc) * Sc;

  // Q fragments for the whole tile (K = HD = 64 -> two 32-deep frags)
  const __bf16* qrow = qbase + (size_t)(qb0 + lm) * HDc + hi8;
  const v16bf qa0 = load_frag16(qrow, 0);
  const v16bf qa1 = load_frag16(qrow, 32);

  float mrow[8], lrow[8];
  v8f acc[4];
#pragma unroll
  for (int r = 0; r < 8; ++r) { mrow[r] = -1e30f; lrow[r] = 0.0f; }
#pragma unroll
  for (int ni = 0; ni < 4; ++ni)
#pragma unroll
    for (int r = 0; r < 8; ++r) acc[ni][r] = 0.0f;

  int lo = qb0 - (WIN - 1);
  if (lo < 0) lo = 0;
  lo &= ~31;

  for (int kb = lo; kb <= qb0 + 15; kb += 32) {
    float sc[2][8], bm[8];
#pragma unroll
    for (int r = 0; r < 8; ++r) bm[r] = -1e30f;

    // ---- scores: two 16-key WMMA tiles over this 32-key block ----
#pragma unroll
    for (int t = 0; t < 2; ++t) {
      const __bf16* krow = kbase + (size_t)(kb + t * 16 + lm) * HDc + hi8;
      v16bf kb0 = load_frag16(krow, 0);
      v16bf kb1 = load_frag16(krow, 32);
      v8f c;
#pragma unroll
      for (int r = 0; r < 8; ++r) c[r] = 0.0f;
      c = wmma_bf16(qa0, kb0, c);
      c = wmma_bf16(qa1, kb1, c);

      const int j = kb + t * 16 + lm;  // key index for this lane's column
#pragma unroll
      for (int r = 0; r < 8; ++r) {
        const int i = qb0 + r + hi * 8;  // query index for this element
        float v = c[r] * SCALE;
        if (!((j <= i) && ((i - j) < WIN))) v = -1e30f;
        sc[t][r] = v;
        bm[r] = fmaxf(bm[r], v);
      }
    }

    // ---- row max across the 16 lanes of each half ----
#pragma unroll
    for (int r = 0; r < 8; ++r) {
      float v = bm[r];
      v = fmaxf(v, __shfl_xor(v, 1, 32));
      v = fmaxf(v, __shfl_xor(v, 2, 32));
      v = fmaxf(v, __shfl_xor(v, 4, 32));
      v = fmaxf(v, __shfl_xor(v, 8, 32));
      bm[r] = v;
    }

    float p[2][8], alpha[8], rs[8];
#pragma unroll
    for (int r = 0; r < 8; ++r) {
      const float mnew = fmaxf(mrow[r], bm[r]);
      alpha[r] = __expf(mrow[r] - mnew);
      mrow[r] = mnew;
      const float p0 = __expf(sc[0][r] - mnew);
      const float p1 = __expf(sc[1][r] - mnew);
      p[0][r] = p0; p[1][r] = p1;
      rs[r] = p0 + p1;
    }
#pragma unroll
    for (int r = 0; r < 8; ++r) {
      float v = rs[r];
      v += __shfl_xor(v, 1, 32);
      v += __shfl_xor(v, 2, 32);
      v += __shfl_xor(v, 4, 32);
      v += __shfl_xor(v, 8, 32);
      lrow[r] = lrow[r] * alpha[r] + v;
    }
#pragma unroll
    for (int ni = 0; ni < 4; ++ni)
#pragma unroll
      for (int r = 0; r < 8; ++r) acc[ni][r] *= alpha[r];

    // ---- stage P (C-frag layout) into LDS, reload as A-frag ----
    __syncthreads();  // protect LDS from previous iteration's readers
#pragma unroll
    for (int t = 0; t < 2; ++t)
#pragma unroll
      for (int r = 0; r < 8; ++r)
        ldsp[r + hi * 8][t * 16 + lm] = (__bf16)p[t][r];
    __syncthreads();

    v16bf ap;
    {
      const __bf16* prow = &ldsp[lm][0];
      v8bf a0 = *(const v8bf*)(prow + hi8);
      v8bf a1 = *(const v8bf*)(prow + 16 + hi8);
#pragma unroll
      for (int j2 = 0; j2 < 8; ++j2) { ap[j2] = a0[j2]; ap[j2 + 8] = a1[j2]; }
    }

    // ---- P @ V : 4 N-tiles of 16 over HD=64 (V stored transposed) ----
#pragma unroll
    for (int ni = 0; ni < 4; ++ni) {
      const __bf16* vrow = vbase + (size_t)(ni * 16 + lm) * Sc + kb + hi8;
      v16bf bv = load_frag16(vrow, 0);
      acc[ni] = wmma_bf16(ap, bv, acc[ni]);
    }
  }

  // ---- epilogue: normalize and store ctx token-major [B][S][H] ----
#pragma unroll
  for (int r = 0; r < 8; ++r) {
    const float inv = 1.0f / lrow[r];
    const int token = qb0 + r + hi * 8;
    const size_t base = ((size_t)b * Sc + token) * Hc + (size_t)h * HDc + lm;
#pragma unroll
    for (int ni = 0; ni < 4; ++ni)
      ctxg[base + ni * 16] = (__bf16)(acc[ni][r] * inv);
  }
}

// ---------------------------------------------------------------------------
// Kernel 4: output projection, f32 result. out = ctx @ wo^T + bo
// ---------------------------------------------------------------------------
__global__ __launch_bounds__(32) void oproj_kernel(
    const __bf16* __restrict__ ctxb, const __bf16* __restrict__ wo,
    const float* __restrict__ bo, float* __restrict__ out) {
  const int lane = threadIdx.x;
  const int lm   = lane & 15;
  const int hi   = lane >> 4;
  const int hi8  = hi << 3;
  const int m0   = blockIdx.x * 16;
  const int n0   = blockIdx.y * 16;

  const __bf16* arow = ctxb + (size_t)(m0 + lm) * Hc + hi8;
  const __bf16* brow = wo   + (size_t)(n0 + lm) * Hc + hi8;

  v8f acc;
#pragma unroll
  for (int r = 0; r < 8; ++r) acc[r] = 0.0f;

#pragma unroll 4
  for (int kk = 0; kk < Hc; kk += 32) {
    v16bf a = load_frag16(arow, kk);
    v16bf b = load_frag16(brow, kk);
    acc = wmma_bf16(a, b, acc);
  }

  const float bn = bo[n0 + lm];
#pragma unroll
  for (int r = 0; r < 8; ++r)
    out[(size_t)(m0 + r + hi * 8) * Hc + n0 + lm] = acc[r] + bn;
}

// ---------------------------------------------------------------------------
// Host launcher
// ---------------------------------------------------------------------------
extern "C" void kernel_launch(void* const* d_in, const int* in_sizes, int n_in,
                              void* d_out, int out_size, void* d_ws,
                              size_t ws_size, hipStream_t stream) {
  (void)in_sizes; (void)n_in; (void)out_size; (void)ws_size;

  const float*     hs  = (const float*)d_in[0];
  const long long* pos = (const long long*)d_in[1];
  const float* wq = (const float*)d_in[2]; const float* bq = (const float*)d_in[3];
  const float* wk = (const float*)d_in[4]; const float* bk = (const float*)d_in[5];
  const float* wv = (const float*)d_in[6]; const float* bv = (const float*)d_in[7];
  const float* wo = (const float*)d_in[8]; const float* bo = (const float*)d_in[9];
  float* out = (float*)d_out;

  // Workspace layout (bf16 buffers), 48 MB total
  uint8_t* ws = (uint8_t*)d_ws;
  const size_t NX = (size_t)Mrows * Hc;   // 4,194,304 elems
  const size_t NW = (size_t)Hc * Hc;      // 1,048,576 elems
  __bf16* xb   = (__bf16*)(ws);
  __bf16* wqb  = (__bf16*)(ws + 2 * NX);
  __bf16* wkb  = (__bf16*)(ws + 2 * NX + 2 * NW);
  __bf16* wvb  = (__bf16*)(ws + 2 * NX + 4 * NW);
  __bf16* wob  = (__bf16*)(ws + 2 * NX + 6 * NW);
  __bf16* qb   = (__bf16*)(ws + 2 * NX + 8 * NW);
  __bf16* kb   = (__bf16*)(ws + 4 * NX + 8 * NW);
  __bf16* vtb  = (__bf16*)(ws + 6 * NX + 8 * NW);
  __bf16* ctxb = (__bf16*)(ws + 8 * NX + 8 * NW);

  // 0) convert inputs to bf16
  cvt_f32_bf16_kernel<<<(int)(NX / 256), 256, 0, stream>>>(hs, xb, (int)NX);
  cvt_f32_bf16_kernel<<<(int)(NW / 256), 256, 0, stream>>>(wq, wqb, (int)NW);
  cvt_f32_bf16_kernel<<<(int)(NW / 256), 256, 0, stream>>>(wk, wkb, (int)NW);
  cvt_f32_bf16_kernel<<<(int)(NW / 256), 256, 0, stream>>>(wv, wvb, (int)NW);
  cvt_f32_bf16_kernel<<<(int)(NW / 256), 256, 0, stream>>>(wo, wob, (int)NW);

  // 1) fused QKV projection (WMMA)
  qkv_proj_kernel<<<dim3(Mrows / 16, Hc / 16, 3), 32, 0, stream>>>(
      xb, wqb, wkb, wvb, bq, bk, bv, qb, kb, vtb);

  // 2) RoPE on q and k
  const int ropeThreads = Bc * NHc * Sc * (HDc / 2);  // 2^21
  rope_kernel<<<ropeThreads / 256, 256, 0, stream>>>(qb, pos);
  rope_kernel<<<ropeThreads / 256, 256, 0, stream>>>(kb, pos);

  // 3) flash attention (WMMA scores + WMMA P·V, f32 online softmax)
  attn_kernel<<<dim3(Sc / 16, NHc, Bc), 32, 0, stream>>>(qb, kb, vtb, ctxb);

  // 4) output projection (WMMA), f32 out
  oproj_kernel<<<dim3(Mrows / 16, Hc / 16), 32, 0, stream>>>(ctxb, wob, bo, out);
}
